// DynamicPillarBEVSideStream_3427383902684
// MI455X (gfx1250) — matrix operand order repaired and checked by reading, not compiled, not run on hardware
//
#include <hip/hip_runtime.h>
#include <hip/hip_bf16.h>

// ---------------- problem constants (from reference) ----------------
#define B_    2
#define N_    120000
#define GX_   512
#define GY_   512
#define P_    (GX_*GY_)
#define RI_H_ 64
#define RI_W_ 2048
#define FPN_  128
#define XMIN_ (-51.2f)
#define YMIN_ (-51.2f)
#define ZMIN_ (-5.0f)
#define XMAX_ (51.2f)
#define YMAX_ (51.2f)
#define ZMAX_ (3.0f)
#define PXS_  (0.2f)
#define PYS_  (0.2f)
#define ZC_   (-1.0f)      /* 0.5*(ZMIN+ZMAX) */
#define NEGF_ (-1.0e9f)
#define M_PTS (B_*N_)      /* 240000, multiple of 64 */

// ---------------- CDNA5 WMMA types ----------------
typedef __attribute__((ext_vector_type(16))) __bf16 bf16x16;
typedef __attribute__((ext_vector_type(8)))  __bf16 bf16x8;
typedef __attribute__((ext_vector_type(8)))  float  f32x8;
typedef __attribute__((ext_vector_type(4)))  int    i32x4;

// gfx1250 async global->LDS path (ASYNCcnt), guarded so absence falls back cleanly
#if defined(__has_builtin)
#if __has_builtin(__builtin_amdgcn_global_load_async_to_lds_b128)
#define HAVE_ASYNC_LDS 1
#endif
#endif
#ifndef HAVE_ASYNC_LDS
#define HAVE_ASYNC_LDS 0
#endif

// ---------------- device helpers ----------------
__device__ __forceinline__ __bf16 f2bf(float f) {
    union { unsigned short u; __bf16 h; } cv;
    unsigned x = __float_as_uint(f);
    cv.u = (unsigned short)((x + 0x7fffu + ((x >> 16) & 1u)) >> 16); // RNE
    return cv.h;
}
__device__ __forceinline__ unsigned encf(float f) { // order-preserving float->u32
    unsigned b = __float_as_uint(f);
    return (b & 0x80000000u) ? ~b : (b | 0x80000000u);
}
__device__ __forceinline__ float decf(unsigned e) {
    return (e & 0x80000000u) ? __uint_as_float(e & 0x7fffffffu) : __uint_as_float(~e);
}
__device__ __forceinline__ float silu_f(float x) { return x / (1.f + __expf(-x)); }

__device__ __forceinline__ bool point_valid(float x, float y, float z, int i, int cnt) {
    return (i < cnt) && x >= XMIN_ && x < XMAX_ && y >= YMIN_ && y < YMAX_ &&
           z >= ZMIN_ && z < ZMAX_;
}
__device__ __forceinline__ int clampi(int v, int lo, int hi) {
    return v < lo ? lo : (v > hi ? hi : v);
}

// Stage a FULL 32(K) x 32(N) bf16 weight chunk into LDS.
// All K/N tile sizes in this network are exact multiples, so no guards needed.
__device__ __forceinline__ void stage_w_chunk(const __bf16* __restrict__ wb, __bf16* lw,
                                              int kc, int ldN, int n0, int tid) {
#if HAVE_ASYNC_LDS
    // 128 threads x 16B = 2KB chunk, per-lane async DMA into LDS
    int row = tid >> 2, q = tid & 3;
    const __bf16* gp = wb + (size_t)(kc + row) * ldN + n0 + q * 8;
    __bf16* lp = lw + row * 32 + q * 8;
    __builtin_amdgcn_global_load_async_to_lds_b128((i32x4*)gp, (i32x4*)lp, 0, 0);
#else
    for (int t = tid; t < 1024; t += 128) {
        int kk = t >> 5, nn = t & 31;
        lw[t] = wb[(size_t)(kc + kk) * ldN + n0 + nn];
    }
#endif
}
__device__ __forceinline__ void stage_fence() {
#if HAVE_ASYNC_LDS
#if __has_builtin(__builtin_amdgcn_s_wait_asynccnt)
    __builtin_amdgcn_s_wait_asynccnt(0);
#else
    asm volatile("s_wait_asynccnt 0x0" ::: "memory");
#endif
#endif
    __syncthreads();
}

// ---------------- fill kernels ----------------
__global__ void k_fill_f32(float* p, float v, long long n) {
    long long i = (long long)blockIdx.x * blockDim.x + threadIdx.x;
    if (i < n) p[i] = v;
}
__global__ void k_fill_enc(unsigned* p, float v, long long n) {
    long long i = (long long)blockIdx.x * blockDim.x + threadIdx.x;
    if (i < n) p[i] = encf(v);
}

// ---------------- pillar statistics (count + xyz sums) ----------------
__global__ void k_pillar_accum(const float* __restrict__ pts, const int* __restrict__ counts,
                               float* __restrict__ cntsum) {
    int g = blockIdx.x * blockDim.x + threadIdx.x;
    if (g >= M_PTS) return;
    int b = g / N_, i = g - b * N_;
    const float* p = pts + (size_t)g * 4;
    float x = p[0], y = p[1], z = p[2];
    if (!point_valid(x, y, z, i, counts[b])) return;
    int xi = clampi((int)floorf((x - XMIN_) / PXS_), 0, GX_ - 1);
    int yi = clampi((int)floorf((y - YMIN_) / PYS_), 0, GY_ - 1);
    float* cs = cntsum + ((size_t)b * P_ + (size_t)yi * GX_ + xi) * 4;
    atomicAdd(cs + 0, x); atomicAdd(cs + 1, y); atomicAdd(cs + 2, z); atomicAdd(cs + 3, 1.f);
}

// ---------------- per-point 11-dim feature (pad K to 32, bf16) ----------------
__global__ void k_pfn_feat(const float* __restrict__ pts, const int* __restrict__ counts,
                           const float* __restrict__ cntsum, __bf16* __restrict__ A1,
                           int* __restrict__ pid_o, unsigned char* __restrict__ valid_o) {
    int g = blockIdx.x * blockDim.x + threadIdx.x;
    if (g >= M_PTS) return;
    int b = g / N_, i = g - b * N_;
    const float* p = pts + (size_t)g * 4;
    float x = p[0], y = p[1], z = p[2], it = p[3];
    bool v = point_valid(x, y, z, i, counts[b]);
    int xi = clampi((int)floorf((x - XMIN_) / PXS_), 0, GX_ - 1);
    int yi = clampi((int)floorf((y - YMIN_) / PYS_), 0, GY_ - 1);
    int pid = v ? (yi * GX_ + xi) : 0;
    const float* cs = cntsum + ((size_t)b * P_ + pid) * 4;
    float cnt = fmaxf(cs[3], 1.f);
    float mx = cs[0] / cnt, my = cs[1] / cnt, mz = cs[2] / cnt;
    float cx = XMIN_ + (xi + 0.5f) * PXS_;
    float cy = YMIN_ + (yi + 0.5f) * PYS_;
    float f[11] = { x, y, z, it, x - mx, y - my, z - mz,
                    x - cx, y - cy, z - ZC_, sqrtf(x * x + y * y + z * z) };
    __bf16* a = A1 + (size_t)g * 32;
    #pragma unroll
    for (int c = 0; c < 11; c++) a[c] = f2bf(f[c]);
    #pragma unroll
    for (int c = 11; c < 32; c++) a[c] = f2bf(0.f);
    pid_o[g] = pid;
    valid_o[g] = v ? 1 : 0;
}

// ---------------- weight repacks to bf16 ----------------
__global__ void k_repack_kn(const float* __restrict__ w, __bf16* __restrict__ o,
                            int K, int Kpad, int N) {
    int i = blockIdx.x * blockDim.x + threadIdx.x;
    if (i >= Kpad * N) return;
    int k = i / N, n = i - k * N;
    o[i] = f2bf(k < K ? w[(size_t)k * N + n] : 0.f);
}
__global__ void k_repack_oihw(const float* __restrict__ w, __bf16* __restrict__ o,
                              int O, int I, int KH) {
    int tot = O * I * KH * KH;
    int i = blockIdx.x * blockDim.x + threadIdx.x;
    if (i >= tot) return;
    int per = I * KH * KH;
    int oc = i / per, rem = i - oc * per;     // rem = ci*KH*KH + r*KH + s
    o[(size_t)rem * O + oc] = f2bf(w[i]);     // -> [K = I*KH*KH][O]
}

// ---------------- generic bf16 WMMA GEMM:  C(MxN) = A(MxKs) * B(KsxN) ----------------
// Wave tile 16x32 (two accumulators, shared A); 4 waves/block share the LDS weight chunk.
// Requires: Ks % 32 == 0, N % 32 == 0 (true for all uses), M padded/guarded.
__global__ void __launch_bounds__(128)
k_gemm_wmma(const __bf16* __restrict__ A, const __bf16* __restrict__ Bw,
            float* __restrict__ C, int M, int N, int Ks) {
    __shared__ __align__(16) __bf16 lw[32 * 32];
    const int tid = threadIdx.x;
    const int lane = tid & 31, wave = tid >> 5;
    const int l15 = lane & 15, lh = lane >> 4;
    const int m0 = (blockIdx.x * 4 + wave) * 16;
    const int n0 = blockIdx.y * 32;
    const int pm = m0 + l15;
    const bool rok = pm < M;
    const bf16x8* ap = (const bf16x8*)(A + (size_t)(rok ? pm : 0) * Ks);
    bf16x8 zz;
    #pragma unroll
    for (int e = 0; e < 8; e++) zz[e] = f2bf(0.f);

    f32x8 acc0 = {}, acc1 = {};
    for (int kc = 0; kc < Ks; kc += 32) {
        stage_w_chunk(Bw, lw, kc, N, n0, tid);
        stage_fence();
        // A fragment: two aligned 16B vector loads (ISA 7.12.2 16-bit A 16x32 layout)
        bf16x8 alo = rok ? ap[(kc >> 3) + lh] : zz;
        bf16x8 ahi = rok ? ap[(kc >> 3) + 2 + lh] : zz;
        bf16x16 af = __builtin_shufflevector(alo, ahi,
                       0, 1, 2, 3, 4, 5, 6, 7, 8, 9, 10, 11, 12, 13, 14, 15);
        bf16x16 b0, b1;
        #pragma unroll
        for (int e = 0; e < 16; e++) {        // B 32x16: K rows in 8-blocks interleaved by lane half (matches A)
            int r = (e < 8 ? lh * 8 + e : 8 + lh * 8 + e) * 32 + l15;
            b0[e] = lw[r];
            b1[e] = lw[r + 16];
        }
        acc0 = __builtin_amdgcn_wmma_f32_16x16x32_bf16(false, af, false, b0,
                                                       (short)0, acc0, false, false);
        acc1 = __builtin_amdgcn_wmma_f32_16x16x32_bf16(false, af, false, b1,
                                                       (short)0, acc1, false, false);
        __syncthreads();
    }
    #pragma unroll
    for (int r = 0; r < 8; r++) {             // D: row = r + 8*lh, col = l15
        int m = m0 + r + (lh << 3);
        if (m < M) {
            C[(size_t)m * N + n0 + l15] = acc0[r];
            C[(size_t)m * N + n0 + 16 + l15] = acc1[r];
        }
    }
}

// ---------------- implicit-GEMM conv (NCHW, square), raw output ----------------
// Wave tile: 16 output pixels x 32 output channels. Cout % 32 == 0, (Cin*KH*KH) % 32 == 0.
template <int KH>
__global__ void __launch_bounds__(128)
k_conv_wmma(const float* __restrict__ in, const __bf16* __restrict__ wb,
            float* __restrict__ out, int Nb, int Cin, int H, int W,
            int Cout, int stride, int pad, int Ho, int Wo) {
    __shared__ __align__(16) __bf16 lw[32 * 32];
    const int tid = threadIdx.x;
    const int lane = tid & 31, wave = tid >> 5;
    const int l15 = lane & 15, lh = lane >> 4;
    const int Mtot = Nb * Ho * Wo;
    const int m0 = (blockIdx.x * 4 + wave) * 16;
    const int n0 = blockIdx.y * 32;
    const int K = Cin * KH * KH;
    const int pm = m0 + l15;
    const bool rok = pm < Mtot;
    int pn = 0, oy = 0, ox = 0;
    if (rok) { pn = pm / (Ho * Wo); int r = pm - pn * Ho * Wo; oy = r / Wo; ox = r - oy * Wo; }
    const int iy0 = oy * stride - pad, ix0 = ox * stride - pad;

    f32x8 acc0 = {}, acc1 = {};
    for (int kc = 0; kc < K; kc += 32) {
        stage_w_chunk(wb, lw, kc, Cout, n0, tid);
        __builtin_prefetch(wb + (size_t)(kc + 32) * Cout + n0, 0, 1); // global_prefetch_b8
        stage_fence();
        bf16x16 af, b0, b1;
        #pragma unroll
        for (int e = 0; e < 16; e++) {        // on-the-fly im2col gather
            int kk = (e < 8) ? (lh * 8 + e) : (16 + lh * 8 + (e - 8));
            int k = kc + kk;
            float v = 0.f;
            if (rok) {
                int ci = k / (KH * KH);
                int rem = k - ci * KH * KH;
                int rr = rem / KH, ss = rem - rr * KH;
                int iy = iy0 + rr, ix = ix0 + ss;
                if (iy >= 0 && iy < H && ix >= 0 && ix < W)
                    v = in[(((size_t)pn * Cin + ci) * H + iy) * W + ix];
            }
            af[e] = f2bf(v);
        }
        #pragma unroll
        for (int e = 0; e < 16; e++) {
            int r = (e < 8 ? lh * 8 + e : 8 + lh * 8 + e) * 32 + l15;
            b0[e] = lw[r];
            b1[e] = lw[r + 16];
        }
        acc0 = __builtin_amdgcn_wmma_f32_16x16x32_bf16(false, af, false, b0,
                                                       (short)0, acc0, false, false);
        acc1 = __builtin_amdgcn_wmma_f32_16x16x32_bf16(false, af, false, b1,
                                                       (short)0, acc1, false, false);
        __syncthreads();
    }
    #pragma unroll
    for (int r = 0; r < 8; r++) {
        int m = m0 + r + (lh << 3);
        if (m < Mtot) {
            int pn2 = m / (Ho * Wo); int rm = m - pn2 * Ho * Wo;
            int oy2 = rm / Wo, ox2 = rm - oy2 * Wo;
            size_t base = (((size_t)pn2 * Cout) * Ho + oy2) * Wo + ox2;
            out[base + (size_t)(n0 + l15) * Ho * Wo] = acc0[r];
            out[base + (size_t)(n0 + 16 + l15) * Ho * Wo] = acc1[r];
        }
    }
}

// ---------------- LayerNorm + SiLU (+bf16 store for next GEMM, + scatter-max) ----------------
__global__ void k_ln_silu(const float* __restrict__ Cm, const float* __restrict__ g,
                          const float* __restrict__ b, int M, int C,
                          __bf16* Aout, int ldA, int colOff,
                          unsigned* grid, int Cg, const int* __restrict__ pid,
                          const unsigned char* __restrict__ valid) {
    int row = blockIdx.x * blockDim.x + threadIdx.x;
    if (row >= M) return;
    const float* xr = Cm + (size_t)row * C;
    float mean = 0.f;
    for (int c = 0; c < C; c++) mean += xr[c];
    mean /= (float)C;
    float var = 0.f;
    for (int c = 0; c < C; c++) { float d = xr[c] - mean; var += d * d; }
    var /= (float)C;
    float inv = rsqrtf(var + 1e-5f);
    int bt = row / N_;
    int pd = pid[row];
    bool v = valid[row] != 0;
    unsigned* gg = grid ? grid + ((size_t)bt * P_ + pd) * Cg : nullptr;
    for (int c = 0; c < C; c++) {
        float y = silu_f((xr[c] - mean) * inv * g[c] + b[c]);
        if (Aout) Aout[(size_t)row * ldA + colOff + c] = f2bf(y);
        if (gg && v) atomicMax(&gg[c], encf(y));
    }
}

// gather m1[pid] (with NEG->0) into cols 32..63 of the pfn2 A matrix
__global__ void k_m1_gather(const unsigned* __restrict__ grid, const int* __restrict__ pid,
                            __bf16* __restrict__ A2, int M) {
    int row = blockIdx.x * blockDim.x + threadIdx.x;
    if (row >= M) return;
    int bt = row / N_;
    const unsigned* gg = grid + ((size_t)bt * P_ + pid[row]) * 32;
    for (int c = 0; c < 32; c++) {
        float m = decf(gg[c]);
        if (m < -1.0e8f) m = 0.f;
        A2[(size_t)row * 64 + 32 + c] = f2bf(m);
    }
}

// decode bev grid [B][P][64] -> float NCHW [B,64,512,512] (NEG->0)
__global__ void k_bev_final(const unsigned* __restrict__ grid, float* __restrict__ bev,
                            long long n) {
    long long i = (long long)blockIdx.x * blockDim.x + threadIdx.x;
    if (i >= n) return;
    int p = (int)(i % P_); long long t = i / P_;
    int c = (int)(t % 64); int bt = (int)(t / 64);
    float v = decf(grid[((size_t)bt * P_ + p) * 64 + c]);
    bev[i] = (v < -1.0e8f) ? 0.f : v;
}

// ---------------- batch-stat BN ----------------
__global__ void __launch_bounds__(256)
k_bn_stats(const float* __restrict__ x, int Nb, int C, int HW,
           float* __restrict__ mean, float* __restrict__ var) {
    __shared__ float s1[256], s2[256];
    int c = blockIdx.x;
    float a = 0.f, b = 0.f;
    for (int n = 0; n < Nb; n++) {
        const float* xp = x + ((size_t)n * C + c) * HW;
        for (int i = threadIdx.x; i < HW; i += 256) { float v = xp[i]; a += v; b += v * v; }
    }
    s1[threadIdx.x] = a; s2[threadIdx.x] = b;
    __syncthreads();
    for (int s = 128; s > 0; s >>= 1) {
        if (threadIdx.x < s) { s1[threadIdx.x] += s1[threadIdx.x + s]; s2[threadIdx.x] += s2[threadIdx.x + s]; }
        __syncthreads();
    }
    if (threadIdx.x == 0) {
        float cnt = (float)Nb * (float)HW;
        float m = s1[0] / cnt;
        mean[c] = m;
        var[c] = s2[0] / cnt - m * m;
    }
}
__global__ void k_bn_act(const float* __restrict__ x, const float* __restrict__ mv,
                         const float* __restrict__ g, const float* __restrict__ b,
                         const float* __restrict__ res, int dosilu, float* __restrict__ out,
                         int C, int HW, long long n) {
    long long i = (long long)blockIdx.x * blockDim.x + threadIdx.x;
    if (i >= n) return;
    int c = (int)((i / HW) % C);
    float y = (x[i] - mv[c]) * rsqrtf(mv[C + c] + 1e-5f) * g[c] + b[c];
    if (res) y += res[i];
    if (dosilu) y = silu_f(y);
    out[i] = y;
}

// ---------------- FPN nearest x2 upsample-add ----------------
__global__ void k_upsample_add(float* __restrict__ dst, const float* __restrict__ src,
                               int C, int H, int W, long long n) {
    long long i = (long long)blockIdx.x * blockDim.x + threadIdx.x;
    if (i >= n) return;
    int x = (int)(i % W); long long t = i / W;
    int y = (int)(t % H); t /= H;
    int c = (int)(t % C); int nb = (int)(t / C);
    dst[i] += src[(((size_t)nb * C + c) * (H / 2) + (y >> 1)) * (W / 2) + (x >> 1)];
}

// ---------------- range-image sampling ----------------
__global__ void k_sample(const float* __restrict__ feat, int sh, int sw,
                         const float* __restrict__ rxyz, const unsigned char* __restrict__ rvalid,
                         float* __restrict__ out, int th, int tw, float cs, long long n) {
    long long i = (long long)blockIdx.x * blockDim.x + threadIdx.x;
    if (i >= n) return;
    int tx = (int)(i % tw); long long t = i / tw;
    int ty = (int)(t % th); t /= th;
    int c = (int)(t % FPN_); int bt = (int)(t / FPN_);
    int ry = ty * (RI_H_ / th), rx = tx * (RI_W_ / tw);
    size_t rb = (size_t)bt * 3 * RI_H_ * RI_W_ + (size_t)ry * RI_W_ + rx;
    float X = rxyz[rb];
    float Y = rxyz[rb + (size_t)RI_H_ * RI_W_];
    bool v = rvalid[(size_t)bt * RI_H_ * RI_W_ + (size_t)ry * RI_W_ + rx] != 0;
    int xi = (int)floorf((X - XMIN_) / cs);
    int yi = (int)floorf((Y - YMIN_) / cs);
    bool ok = v && __builtin_isfinite(X) && __builtin_isfinite(Y) &&
              xi >= 0 && xi < sw && yi >= 0 && yi < sh;
    int cy = clampi(yi, 0, sh - 1), cx = clampi(xi, 0, sw - 1);
    out[i] = ok ? feat[((size_t)bt * FPN_ + c) * ((size_t)sh * sw) + (size_t)cy * sw + cx] : 0.f;
}

// =====================================================================
//                              host side
// =====================================================================
extern "C" void kernel_launch(void* const* d_in, const int* in_sizes, int n_in,
                              void* d_out, int out_size, void* d_ws, size_t ws_size,
                              hipStream_t stream) {
    (void)in_sizes; (void)n_in; (void)out_size; (void)ws_size;
    // ---- inputs (setup_inputs() insertion order; params flattened in insertion order) ----
    const float* raw_points = (const float*)d_in[0];
    const int*   counts     = (const int*)d_in[1];
    const float* rxyz       = (const float*)d_in[2];
    const unsigned char* rvalid = (const unsigned char*)d_in[3];
    auto F = [&](int i) { return (const float*)d_in[i]; };
    const int b0_idx[3] = { 10, 25, 40 };
    const int b1_idx[3] = { 19, 34, 49 };
    const int s_cin[3]  = { 64, 64, 128 };
    const int s_cout[3] = { 64, 128, 256 };

    // ---- workspace arena ----
    char* wsb = (char*)d_ws;
    size_t off = 0;
    auto alloc = [&](size_t bytes) -> void* {
        void* p = wsb + off;
        off = (off + bytes + 255) & ~(size_t)255;
        return p;
    };
    auto gsz = [](long long n, int bs) { return (unsigned)((n + bs - 1) / bs); };

    float*    cntsum  = (float*)alloc((size_t)B_ * P_ * 4 * 4);
    int*      pid     = (int*)alloc((size_t)M_PTS * 4);
    unsigned char* vb = (unsigned char*)alloc((size_t)M_PTS);
    __bf16*   A1      = (__bf16*)alloc((size_t)M_PTS * 32 * 2);
    float*    C1      = (float*)alloc((size_t)M_PTS * 32 * 4);
    unsigned* m1grid  = (unsigned*)alloc((size_t)B_ * P_ * 32 * 4);
    __bf16*   A2      = (__bf16*)alloc((size_t)M_PTS * 64 * 2);
    float*    C2      = (float*)alloc((size_t)M_PTS * 64 * 4);
    unsigned* bevgrid = (unsigned*)alloc((size_t)B_ * P_ * 64 * 4);
    float*    bev     = (float*)alloc((size_t)B_ * 64 * P_ * 4);
    float*    mv      = (float*)alloc(2 * 256 * 4);
    __bf16*   w1b     = (__bf16*)alloc(32 * 32 * 2);
    __bf16*   w2b     = (__bf16*)alloc(64 * 64 * 2);

    auto repack = [&](const float* w, int O, int I, int KH) -> __bf16* {
        int tot = O * I * KH * KH;
        __bf16* o = (__bf16*)alloc((size_t)tot * 2);
        k_repack_oihw<<<gsz(tot, 256), 256, 0, stream>>>(w, o, O, I, KH);
        return o;
    };

    // ---- PFN ----
    k_fill_f32<<<gsz((long long)B_ * P_ * 4, 256), 256, 0, stream>>>(cntsum, 0.f, (long long)B_ * P_ * 4);
    k_fill_enc<<<gsz((long long)B_ * P_ * 32, 256), 256, 0, stream>>>(m1grid, NEGF_, (long long)B_ * P_ * 32);
    k_fill_enc<<<gsz((long long)B_ * P_ * 64, 256), 256, 0, stream>>>(bevgrid, NEGF_, (long long)B_ * P_ * 64);
    k_pillar_accum<<<gsz(M_PTS, 256), 256, 0, stream>>>(raw_points, counts, cntsum);
    k_pfn_feat<<<gsz(M_PTS, 256), 256, 0, stream>>>(raw_points, counts, cntsum, A1, pid, vb);
    k_repack_kn<<<gsz(32 * 32, 256), 256, 0, stream>>>(F(4), w1b, 11, 32, 32);
    k_repack_kn<<<gsz(64 * 64, 256), 256, 0, stream>>>(F(7), w2b, 64, 64, 64);

    k_gemm_wmma<<<dim3(M_PTS / 64, 1), 128, 0, stream>>>(A1, w1b, C1, M_PTS, 32, 32);
    k_ln_silu<<<gsz(M_PTS, 256), 256, 0, stream>>>(C1, F(5), F(6), M_PTS, 32,
                                                   A2, 64, 0, m1grid, 32, pid, vb);
    k_m1_gather<<<gsz(M_PTS, 256), 256, 0, stream>>>(m1grid, pid, A2, M_PTS);
    k_gemm_wmma<<<dim3(M_PTS / 64, 2), 128, 0, stream>>>(A2, w2b, C2, M_PTS, 64, 64);
    k_ln_silu<<<gsz(M_PTS, 256), 256, 0, stream>>>(C2, F(8), F(9), M_PTS, 64,
                                                   nullptr, 0, 0, bevgrid, 64, pid, vb);
    k_bev_final<<<gsz((long long)B_ * 64 * P_, 256), 256, 0, stream>>>(
        bevgrid, bev, (long long)B_ * 64 * P_);

    // ---- conv helpers ----
    auto conv3 = [&](const float* in, const __bf16* w, float* out,
                     int Cin, int H, int Cout, int stride, int Ho) {
        dim3 g(gsz((long long)B_ * Ho * Ho, 64), Cout / 32);
        k_conv_wmma<3><<<g, 128, 0, stream>>>(in, w, out, B_, Cin, H, H, Cout, stride, 1, Ho, Ho);
    };
    auto conv1 = [&](const float* in, const __bf16* w, float* out,
                     int Cin, int H, int Cout, int stride, int Ho) {
        dim3 g(gsz((long long)B_ * Ho * Ho, 64), Cout / 32);
        k_conv_wmma<1><<<g, 128, 0, stream>>>(in, w, out, B_, Cin, H, H, Cout, stride, 0, Ho, Ho);
    };
    auto bnstats = [&](const float* x, int C, int HW) {
        k_bn_stats<<<C, 256, 0, stream>>>(x, B_, C, HW, mv, mv + C);
    };
    auto bnact = [&](const float* x, const float* g, const float* b, const float* res,
                     int dosilu, float* out, int C, int HW) {
        long long n = (long long)B_ * C * HW;
        k_bn_act<<<gsz(n, 256), 256, 0, stream>>>(x, mv, g, b, res, dosilu, out, C, HW, n);
    };

    auto resblock = [&](const float* x, int cin, int Hi, int cout, int stride,
                        const __bf16* w1, const float* g1, const float* b1,
                        const __bf16* w2, const float* g2, const float* b2,
                        const __bf16* wd, const float* gd, const float* bd,
                        float* t1, float* t2, float* rd, float* out) {
        int Ho = (Hi + 2 - 3) / stride + 1;
        int HW = Ho * Ho;
        conv3(x, w1, t1, cin, Hi, cout, stride, Ho);
        bnstats(t1, cout, HW);
        bnact(t1, g1, b1, nullptr, 1, t1, cout, HW);
        conv3(t1, w2, t2, cout, Ho, cout, 1, Ho);
        const float* res = x;
        if (wd) {
            conv1(x, wd, rd, cin, Hi, cout, stride, Ho);
            bnstats(rd, cout, HW);
            bnact(rd, gd, bd, nullptr, 0, rd, cout, HW);
            res = rd;
        }
        bnstats(t2, cout, HW);
        bnact(t2, g2, b2, res, 1, out, cout, HW);
    };

    // ---- backbone stages ----
    float* feats[3];
    const float* xin = bev;
    int Hi = 512, cin = 64;
    for (int s = 0; s < 3; s++) {
        int cout = s_cout[s];
        int Ho = Hi / 2;
        size_t sz = (size_t)B_ * cout * Ho * Ho * 4;
        float* t1 = (float*)alloc(sz);
        float* t2 = (float*)alloc(sz);
        float* rd = (float*)alloc(sz);
        float* y0 = (float*)alloc(sz);
        float* ft = (float*)alloc(sz);
        int i0 = b0_idx[s], i1 = b1_idx[s];
        __bf16* wb01 = repack(F(i0 + 0), cout, s_cin[s], 3);
        __bf16* wb02 = repack(F(i0 + 3), cout, cout, 3);
        __bf16* wb0d = repack(F(i0 + 6), cout, s_cin[s], 1);
        __bf16* wb11 = repack(F(i1 + 0), cout, cout, 3);
        __bf16* wb12 = repack(F(i1 + 3), cout, cout, 3);
        resblock(xin, cin, Hi, cout, 2, wb01, F(i0 + 1), F(i0 + 2),
                 wb02, F(i0 + 4), F(i0 + 5), wb0d, F(i0 + 7), F(i0 + 8), t1, t2, rd, y0);
        resblock(y0, cout, Ho, cout, 1, wb11, F(i1 + 1), F(i1 + 2),
                 wb12, F(i1 + 4), F(i1 + 5), nullptr, nullptr, nullptr, t1, t2, rd, ft);
        feats[s] = ft;
        xin = ft; Hi = Ho; cin = cout;
    }

    // ---- FPN ----
    const int fH[3] = { 256, 128, 64 };
    float* lat[3]; float* fsm[3]; float* ftm[3];
    __bf16* latw[3]; __bf16* smw[3];
    for (int s = 0; s < 3; s++) {
        size_t sz = (size_t)B_ * FPN_ * fH[s] * fH[s] * 4;
        lat[s] = (float*)alloc(sz);
        fsm[s] = (float*)alloc(sz);
        ftm[s] = (float*)alloc(sz);
        latw[s] = repack(F(55 + s), FPN_, s_cout[s], 1);
        smw[s]  = repack(F(58 + 3 * s), FPN_, FPN_, 3);
    }
    auto smooth = [&](int s) {
        int Hs = fH[s], HW = Hs * Hs;
        conv3(lat[s], smw[s], ftm[s], FPN_, Hs, FPN_, 1, Hs);
        bnstats(ftm[s], FPN_, HW);
        bnact(ftm[s], F(58 + 3 * s + 1), F(58 + 3 * s + 2), nullptr, 1, fsm[s], FPN_, HW);
    };
    conv1(feats[2], latw[2], lat[2], 256, 64, FPN_, 1, 64);
    smooth(2);
    conv1(feats[1], latw[1], lat[1], 128, 128, FPN_, 1, 128);
    {
        long long n = (long long)B_ * FPN_ * 128 * 128;
        k_upsample_add<<<gsz(n, 256), 256, 0, stream>>>(lat[1], lat[2], FPN_, 128, 128, n);
    }
    smooth(1);
    conv1(feats[0], latw[0], lat[0], 64, 256, FPN_, 1, 256);
    {
        long long n = (long long)B_ * FPN_ * 256 * 256;
        k_upsample_add<<<gsz(n, 256), 256, 0, stream>>>(lat[0], lat[1], FPN_, 256, 256, n);
    }
    smooth(0);

    // ---- range-image sampling to outputs ----
    float* out0 = (float*)d_out;
    float* out1 = out0 + (size_t)B_ * FPN_ * 32 * 1024;
    float* out2 = out1 + (size_t)B_ * FPN_ * 16 * 512;
    {
        long long n = (long long)B_ * FPN_ * 32 * 1024;
        k_sample<<<gsz(n, 256), 256, 0, stream>>>(fsm[0], 256, 256, rxyz, rvalid, out0, 32, 1024, 0.4f, n);
    }
    {
        long long n = (long long)B_ * FPN_ * 16 * 512;
        k_sample<<<gsz(n, 256), 256, 0, stream>>>(fsm[1], 128, 128, rxyz, rvalid, out1, 16, 512, 0.8f, n);
    }
    {
        long long n = (long long)B_ * FPN_ * 8 * 256;
        k_sample<<<gsz(n, 256), 256, 0, stream>>>(fsm[2], 64, 64, rxyz, rvalid, out2, 8, 256, 1.6f, n);
    }
}
